// TransE_23021024706777
// MI455X (gfx1250) — compile-verified, matrix operand order checked by reading, not run on hardware
//
#include <hip/hip_runtime.h>
#include <hip/hip_bf16.h>
#include <math.h>
#include <stdint.h>

typedef __bf16 v16bf __attribute__((ext_vector_type(16)));
typedef float  v8f   __attribute__((ext_vector_type(8)));

#define N_ENT   100000
#define DIM     128
#define NB      16
#define EPSF    1e-12f
#define TILE    256     // entities per block
#define ROWPAD  132     // padded LDS row stride in dwords (16B-aligned, breaks bank aliasing)

// ---------------------------------------------------------------------------
// Kernel A: head[b][d] = normalize(ent[e1[b]])[d] + normalize(rel[rel[b]])[d]
// 16 blocks x 128 threads
// ---------------------------------------------------------------------------
__global__ __launch_bounds__(128) void head_prep(const int* __restrict__ e1,
                                                 const int* __restrict__ rl,
                                                 const float* __restrict__ ent,
                                                 const float* __restrict__ relE,
                                                 float* __restrict__ head) {
  __shared__ float red[128];
  const int b = blockIdx.x;
  const int d = threadIdx.x;
  const float ae = ent [(size_t)e1[b] * DIM + d];
  const float re = relE[(size_t)rl[b] * DIM + d];

  red[d] = ae * ae; __syncthreads();
  for (int s = 64; s > 0; s >>= 1) { if (d < s) red[d] += red[d + s]; __syncthreads(); }
  const float na = sqrtf(red[0]); __syncthreads();

  red[d] = re * re; __syncthreads();
  for (int s = 64; s > 0; s >>= 1) { if (d < s) red[d] += red[d + s]; __syncthreads(); }
  const float nr = sqrtf(red[0]);

  head[b * DIM + d] = ae / fmaxf(na, EPSF) + re / fmaxf(nr, EPSF);
}

// ---------------------------------------------------------------------------
// Kernel B: dist[b][n] = sum_d |head[b][d] - ent[n][d]/||ent[n]|||
// grid = ceil(N/TILE) blocks x 256 threads. Lane owns one entity.
//  - tile staged global->LDS via CDNA5 async loads (ASYNCcnt)
//  - per-entity sum-of-squares reduced on the matrix pipe (v_wmma bf16, A=ones)
// ---------------------------------------------------------------------------
__global__ __launch_bounds__(256) void dist_kernel(const float* __restrict__ ent,
                                                   const float* __restrict__ head_g,
                                                   float* __restrict__ out) {
  extern __shared__ float lds[];
  float* tile  = lds;                       // TILE * ROWPAD dwords
  float* headS = lds + TILE * ROWPAD;       // NB * DIM
  float* rnorm = headS + NB * DIM;          // TILE

  const int tid  = threadIdx.x;
  const int base = blockIdx.x * TILE;

  // head -> LDS (tiny, plain loads)
  for (int i = tid; i < NB * DIM; i += 256) headS[i] = head_g[i];

  // Async-stage entity tile: 256 rows x 128 f32 = 8192 x b128 chunks.
  // Dynamic LDS base is 0 (no static __shared__ in this kernel), so raw
  // offsets are valid LDS addresses for the async-load VDST operand.
  for (int it = 0; it < 32; ++it) {
    const int id  = tid + 256 * it;          // 0..8191
    const int row = id >> 5;                 // 32 chunks per row
    const int cir = id & 31;
    int grow = base + row; if (grow >= N_ENT) grow = N_ENT - 1;   // clamp tail
    const unsigned lds_off = (unsigned)((row * ROWPAD + cir * 4) * 4);
    const unsigned long long ga =
        (unsigned long long)(uintptr_t)(ent + (size_t)grow * DIM + cir * 4);
    asm volatile("global_load_async_to_lds_b128 %0, %1, off"
                 :: "v"(lds_off), "v"(ga));
  }
  asm volatile("s_wait_asynccnt 0" ::: "memory");
  __syncthreads();

  // --- WMMA sumsq pre-pass: rnorm[n] = 1/max(sqrt(sum_d x^2), eps) ---------
  // D = ones(16x32) * B, with B[k][n] = x^2 for entity (g*16 + n), d-chunk k.
  // Lane L supplies column n = L%16, K-half = L/16 (16 bf16 values / chunk).
  const int lane = tid & 31;
  const int wave = tid >> 5;                 // 8 waves
  v16bf ones;
  #pragma unroll
  for (int i = 0; i < 16; ++i) ones[i] = (__bf16)1.0f;

  for (int g = wave; g < TILE / 16; g += 8) {       // 2 groups per wave
    const int r  = g * 16 + (lane & 15);
    const int kh = lane >> 4;
    v8f acc = {};
    #pragma unroll
    for (int c = 0; c < 4; ++c) {                   // 4 x K=32 covers D=128
      // 16B-aligned: row stride 132 dw (528B = 33*16B), col offset multiple of 16 dw
      const float4* p4 = (const float4*)__builtin_assume_aligned(
          tile + r * ROWPAD + c * 32 + kh * 16, 16);
      v16bf bm;
      #pragma unroll
      for (int q = 0; q < 4; ++q) {                 // 4 x ds_load_b128
        const float4 v = p4[q];
        bm[q * 4 + 0] = (__bf16)(v.x * v.x);
        bm[q * 4 + 1] = (__bf16)(v.y * v.y);
        bm[q * 4 + 2] = (__bf16)(v.z * v.z);
        bm[q * 4 + 3] = (__bf16)(v.w * v.w);
      }
      acc = __builtin_amdgcn_wmma_f32_16x16x32_bf16(false, ones, false, bm,
                                                    (short)0, acc, false, false);
    }
    if (lane < 16) {                                // D[0][n] lives in lanes 0..15
      rnorm[g * 16 + lane] = 1.0f / fmaxf(sqrtf(acc[0]), EPSF);
    }
  }
  __syncthreads();

  // --- main L1-distance loop: lane owns entity row `tid` -------------------
  float acc[NB];
  #pragma unroll
  for (int b = 0; b < NB; ++b) acc[b] = 0.0f;
  const float ri = rnorm[tid];
  const float4* __restrict__ myrow4 =
      (const float4*)__builtin_assume_aligned(tile + tid * ROWPAD, 16);  // 33 float4/row
  const float4* __restrict__ head4 =
      (const float4*)__builtin_assume_aligned(headS, 16);                // NB*32 float4

  for (int c = 0; c < 32; ++c) {
    const float4 x4 = myrow4[c];                    // ds_load_b128
    const float x0 = x4.x * ri, x1 = x4.y * ri, x2 = x4.z * ri, x3 = x4.w * ri;
    #pragma unroll
    for (int b = 0; b < NB; ++b) {
      const float4 h = head4[b * 32 + c];           // ds_load_b128 broadcast
      acc[b] += __builtin_fabsf(h.x - x0) + __builtin_fabsf(h.y - x1)
              + __builtin_fabsf(h.z - x2) + __builtin_fabsf(h.w - x3);
    }
  }

  const int n = base + tid;
  if (n < N_ENT) {
    #pragma unroll
    for (int b = 0; b < NB; ++b) out[(size_t)b * N_ENT + n] = acc[b];  // coalesced
  }
}

// ---------------------------------------------------------------------------
// Kernel C: deterministic in-place row softmax over N_ENT (16 blocks x 1024)
// ---------------------------------------------------------------------------
__global__ __launch_bounds__(1024) void softmax_rows(float* __restrict__ data) {
  __shared__ float red[1024];
  float* p = data + (size_t)blockIdx.x * N_ENT;
  const int t = threadIdx.x;

  float m = -3.4e38f;
  for (int i = t; i < N_ENT; i += 1024) m = fmaxf(m, p[i]);
  red[t] = m; __syncthreads();
  for (int s = 512; s > 0; s >>= 1) { if (t < s) red[t] = fmaxf(red[t], red[t + s]); __syncthreads(); }
  const float rowmax = red[0]; __syncthreads();

  float ssum = 0.0f;
  for (int i = t; i < N_ENT; i += 1024) ssum += __expf(p[i] - rowmax);
  red[t] = ssum; __syncthreads();
  for (int s = 512; s > 0; s >>= 1) { if (t < s) red[t] += red[t + s]; __syncthreads(); }
  const float inv = 1.0f / red[0]; __syncthreads();

  for (int i = t; i < N_ENT; i += 1024) p[i] = __expf(p[i] - rowmax) * inv;
}

// ---------------------------------------------------------------------------
extern "C" void kernel_launch(void* const* d_in, const int* in_sizes, int n_in,
                              void* d_out, int out_size, void* d_ws, size_t ws_size,
                              hipStream_t stream) {
  (void)in_sizes; (void)n_in; (void)out_size; (void)ws_size;
  const int*   e1   = (const int*)d_in[0];
  const int*   rl   = (const int*)d_in[1];
  // d_in[2] (X) and d_in[3] (A) are unused by the reference forward.
  const float* ent  = (const float*)d_in[4];
  const float* relE = (const float*)d_in[5];
  float* head = (float*)d_ws;          // 16*128 f32 = 8 KB
  float* out  = (float*)d_out;         // 16*100000 f32

  head_prep<<<NB, 128, 0, stream>>>(e1, rl, ent, relE, head);

  const size_t shmem = (size_t)(TILE * ROWPAD + NB * DIM + TILE) * sizeof(float);
  (void)hipFuncSetAttribute(reinterpret_cast<const void*>(dist_kernel),
                            hipFuncAttributeMaxDynamicSharedMemorySize, (int)shmem);
  const int nblk = (N_ENT + TILE - 1) / TILE;
  dist_kernel<<<nblk, 256, shmem, stream>>>(ent, head, out);

  softmax_rows<<<NB, 1024, 0, stream>>>(out);
}
